// Net_88484916232752
// MI455X (gfx1250) — compile-verified, matrix operand order checked by reading, not run on hardware
//
#include <hip/hip_runtime.h>
#include <hip/hip_bf16.h>

typedef __attribute__((ext_vector_type(16))) _Float16 v16h;
typedef __attribute__((ext_vector_type(8)))  float    v8f;
typedef __attribute__((ext_vector_type(4)))  unsigned int v4u;
typedef __attribute__((ext_vector_type(8)))  int      v8i;
typedef __attribute__((ext_vector_type(4)))  int      v4i;

#define WMMA16(a, b, c) __builtin_amdgcn_wmma_f32_16x16x32_f16( \
    false, (a), false, (b), (short)0, (c), false, false)

#define BATCH 4096
#define TT 32
#define DD 256
#define HID 32

__device__ __forceinline__ float sigm(float x) {
    return 1.0f / (1.0f + __expf(-x));
}

// ---------------------------------------------------------------------------
// Kernel 0: temporal adjacency adj_t = norm(exp(-|i-j|) + I), [32x32] f32
// ---------------------------------------------------------------------------
__global__ __launch_bounds__(32) void adjt_kernel(float* __restrict__ adjt) {
    __shared__ float a[TT][TT];
    __shared__ float dinv[TT];
    int i = threadIdx.x;
    float s = 0.0f;
    for (int j = 0; j < TT; ++j) {
        int d = i > j ? i - j : j - i;
        float v = __expf(-(float)d) + (i == j ? 1.0f : 0.0f);
        a[i][j] = v;
        s += v;
    }
    dinv[i] = rsqrtf(s);
    __syncthreads();
    for (int j = 0; j < TT; ++j)
        adjt[i * TT + j] = a[i][j] * dinv[i] * dinv[j];
}

// ---------------------------------------------------------------------------
// Kernel 0b: transpose + f16-convert GCN weights: w1T[n][k] = w1[k][n]
// ---------------------------------------------------------------------------
__global__ __launch_bounds__(256) void prep_w_kernel(const float* __restrict__ w1,
                                                     const float* __restrict__ w3,
                                                     _Float16* __restrict__ w1T,
                                                     _Float16* __restrict__ w3T) {
    int idx = blockIdx.x * 256 + threadIdx.x;   // 0 .. 256*128-1
    if (idx < DD * 128) {
        int k = idx >> 7;       // 0..255
        int n = idx & 127;      // 0..127
        w1T[n * DD + k] = (_Float16)w1[idx];
        w3T[n * DD + k] = (_Float16)w3[idx];
    }
}

// ---------------------------------------------------------------------------
// TDM helper: stage a 2-D f32 tile [rows x cols] (row stride = cols) from
// global memory into LDS via TENSOR_LOAD_TO_LDS (TENSORcnt-tracked DMA).
// Call from ONE wave only (TDM ignores EXEC; issues once per wave).
// D# layout per CDNA5 ISA ch.8: group0 = {flags, lds_addr, global_addr, type},
// group1 = {mask/data_size, dims, tile dims, strides}, groups 2/3 zero (2-D).
// ---------------------------------------------------------------------------
__device__ __forceinline__ void tdm_load_tile_f32(const float* gsrc, float* lds_dst,
                                                  int cols, int rows) {
    unsigned long long ga = (unsigned long long)(size_t)gsrc;
    unsigned int lds_off = (unsigned int)(size_t)lds_dst;  // low 32 bits = LDS offset

    v4u g0;
    g0[0] = 1u;                                        // count=1, user mode, no gather
    g0[1] = lds_off;                                   // lds_addr
    g0[2] = (unsigned int)(ga & 0xFFFFFFFFu);          // global_addr[31:0]
    g0[3] = (unsigned int)((ga >> 32) & 0x01FFFFFFu)   // global_addr[56:32]
          | (2u << 30);                                // type=2 ("image")

    v8i g1;
    g1[0] = (2 << 16);                                 // workgroup_mask=0, data_size=4B
    g1[1] = (int)((unsigned)(cols & 0xFFFF) << 16);    // tensor_dim0[15:0]
    g1[2] = (int)(((unsigned)cols >> 16) | ((unsigned)(rows & 0xFFFF) << 16));
    g1[3] = (int)(((unsigned)rows >> 16) | ((unsigned)(cols & 0xFFFF) << 16)); // tile_dim0
    g1[4] = rows;                                      // tile_dim1 (tile_dim2=0)
    g1[5] = cols;                                      // tensor_dim0_stride[31:0]
    g1[6] = 0;
    g1[7] = 0;

    v4i gz = {0, 0, 0, 0};
#if __clang_major__ >= 23
    v8i gz8 = {0, 0, 0, 0, 0, 0, 0, 0};
    __builtin_amdgcn_tensor_load_to_lds(g0, g1, gz, gz, gz8, 0);
#else
    __builtin_amdgcn_tensor_load_to_lds(g0, g1, gz, gz, 0);
#endif
    __builtin_amdgcn_s_wait_tensorcnt(0);
}

// ---------------------------------------------------------------------------
// Kernel 1: fused adjacency + GCN, one block per batch sample
//   out: hbuf[b][t][128] f16 = relu(adj_f @ (x@w1+b1) + adj_t @ (x@w3+b3))
// ---------------------------------------------------------------------------
__global__ __launch_bounds__(256) void gcn_kernel(const float* __restrict__ x,
                                                  const float* __restrict__ ls,
                                                  const float* __restrict__ b1v,
                                                  const float* __restrict__ b3v,
                                                  const float* __restrict__ adjt_g,
                                                  const _Float16* __restrict__ w1T,
                                                  const _Float16* __restrict__ w3T,
                                                  _Float16* __restrict__ hbuf) {
    __shared__ alignas(16) float    xstage[TT * DD];   // TDM-staged x tile (32 KB)
    __shared__ alignas(32) _Float16 xh[TT][DD];        // x as f16
    __shared__ alignas(32) _Float16 xsh[TT][DD];       // x/ls as f16
    __shared__ float sq[TT];
    __shared__ float gram[TT][TT];
    __shared__ float adjf[TT][TT];
    __shared__ alignas(32) _Float16 adjh[TT][TT];
    __shared__ alignas(32) _Float16 adjth[TT][TT];
    __shared__ alignas(32) _Float16 xw1T[128][TT];     // (x@w1+b1)^T f16
    __shared__ alignas(32) _Float16 xw3T[128][TT];

    const int tid  = threadIdx.x;
    const int lane = tid & 31;
    const int wave = tid >> 5;          // 0..7

    // ---- phase 0: TDM stage x[b] tile into LDS (wave 0 issues the DMA) ----
    const float* xb = x + (size_t)blockIdx.x * TT * DD;
    if (wave == 0) {
        tdm_load_tile_f32(xb, xstage, DD, TT);
    }
    if (tid < TT) sq[tid] = 0.0f;
    __syncthreads();                    // staged tile + sq visible to all waves

    // ---- phase 1: build f16 copies + ARD row norms from staged tile ----
    {
        int row = tid >> 3;             // 32 rows, 8 threads each
        int c0  = (tid & 7) * 32;       // 32 contiguous cols
        float acc = 0.0f;
        for (int c = 0; c < 32; ++c) {
            float v = xstage[row * DD + c0 + c];
            float s = v / ls[c0 + c];
            xh[row][c0 + c]  = (_Float16)v;
            xsh[row][c0 + c] = (_Float16)s;
            acc += s * s;
        }
        atomicAdd(&sq[row], acc);       // ds_add_f32
    }
    __syncthreads();

    // ---- phase 2: Gram = xs @ xs^T via WMMA (waves 0..3, 2x2 tiles) ----
    if (wave < 4) {
        const int mt = wave >> 1, nt = wave & 1;
        const int m  = mt * 16 + (lane & 15);
        const int n  = nt * 16 + (lane & 15);
        const int kh = (lane >> 4) * 16;
        v8f c = {};
        for (int k0 = 0; k0 < DD; k0 += 32) {
            v16h a = *(const v16h*)&xsh[m][k0 + kh];
            v16h b = *(const v16h*)&xsh[n][k0 + kh];
            c = WMMA16(a, b, c);
        }
        const int ccol  = nt * 16 + (lane & 15);
        const int rbase = mt * 16 + ((lane >> 4) ? 8 : 0);
        for (int r = 0; r < 8; ++r) gram[rbase + r][ccol] = c[r];
    }
    __syncthreads();

    // ---- phase 3: adjacency (RBF + I, symmetric normalization) ----
    for (int e = tid; e < TT * TT; e += 256) {
        int i = e >> 5, j = e & 31;
        float d2 = sq[i] + sq[j] - 2.0f * gram[i][j];
        d2 = fmaxf(d2, 0.0f);
        adjf[i][j] = __expf(-0.5f * d2) + (i == j ? 1.0f : 0.0f);
    }
    __syncthreads();
    if (tid < TT) {
        float s = 0.0f;
        for (int j = 0; j < TT; ++j) s += adjf[tid][j];
        sq[tid] = rsqrtf(s);            // reuse sq as D^-1/2
    }
    __syncthreads();
    for (int e = tid; e < TT * TT; e += 256) {
        int i = e >> 5, j = e & 31;
        adjh[i][j]  = (_Float16)(adjf[i][j] * sq[i] * sq[j]);
        adjth[i][j] = (_Float16)adjt_g[e];
    }
    __syncthreads();

    // ---- phase 4: xw1 = x@w1+b1, xw3 = x@w3+b3; store transposed f16 ----
    {
        const int nt   = wave;                       // N tile 0..7
        const int ncol = nt * 16 + (lane & 15);
        const int kh   = (lane >> 4) * 16;
        const _Float16* w1row = w1T + (size_t)ncol * DD;
        const _Float16* w3row = w3T + (size_t)ncol * DD;
        for (int mt = 0; mt < 2; ++mt) {
            const int m = mt * 16 + (lane & 15);
            v8f c1, c3;
            float bb1 = b1v[ncol], bb3 = b3v[ncol];
            for (int r = 0; r < 8; ++r) { c1[r] = bb1; c3[r] = bb3; }
            for (int k0 = 0; k0 < DD; k0 += 32) {
                v16h a  = *(const v16h*)&xh[m][k0 + kh];
                v16h b1 = *(const v16h*)(w1row + k0 + kh);
                v16h b3 = *(const v16h*)(w3row + k0 + kh);
                c1 = WMMA16(a, b1, c1);
                c3 = WMMA16(a, b3, c3);
            }
            const int rbase = mt * 16 + ((lane >> 4) ? 8 : 0);
            for (int r = 0; r < 8; ++r) {
                xw1T[ncol][rbase + r] = (_Float16)c1[r];
                xw3T[ncol][rbase + r] = (_Float16)c3[r];
            }
        }
    }
    __syncthreads();

    // ---- phase 5: h = relu(adj_f @ xw1 + adj_t @ xw3), K=32 ----
    {
        const int nt   = wave;
        const int ncol = nt * 16 + (lane & 15);
        const int kh   = (lane >> 4) * 16;
        _Float16* hb = hbuf + (size_t)blockIdx.x * TT * 128;
        for (int mt = 0; mt < 2; ++mt) {
            const int m = mt * 16 + (lane & 15);
            v8f c = {};
            v16h af = *(const v16h*)&adjh[m][kh];
            v16h bf = *(const v16h*)&xw1T[ncol][kh];
            c = WMMA16(af, bf, c);
            v16h at = *(const v16h*)&adjth[m][kh];
            v16h bt = *(const v16h*)&xw3T[ncol][kh];
            c = WMMA16(at, bt, c);
            const int rbase = mt * 16 + ((lane >> 4) ? 8 : 0);
            for (int r = 0; r < 8; ++r)
                hb[(size_t)(rbase + r) * 128 + ncol] = (_Float16)fmaxf(c[r], 0.0f);
        }
    }
}

// ---------------------------------------------------------------------------
// Kernel 2: one LSTM layer.  256 blocks x 64 threads.
//   Block owns 16 batch rows; wave w owns hid slice [16w,16w+16).
//   Per timestep: g = bias + x_t@Wi^T + h@Wh^T  via WMMA, then gate math.
// ---------------------------------------------------------------------------
__global__ __launch_bounds__(64) void lstm_kernel(const _Float16* __restrict__ in_seq,
                                                  _Float16* __restrict__ out_seq,
                                                  const float* __restrict__ Wi,
                                                  const float* __restrict__ Wh,
                                                  const float* __restrict__ bi,
                                                  const float* __restrict__ bh,
                                                  int in_dim,
                                                  float* __restrict__ hfinal) {
    __shared__ alignas(32) _Float16 Wih[128 * 128]; // [gate_n][k], row-major (= Wi layout)
    __shared__ alignas(32) _Float16 Whh[128 * HID];
    __shared__ float bsum[128];
    __shared__ alignas(32) _Float16 htmp[16][HID];  // h in A-fragment (row-major) layout

    const int tid  = threadIdx.x;
    const int lane = tid & 31;
    const int w    = tid >> 5;              // 0..1: hid half

    for (int i = tid; i < 128 * in_dim; i += 64) Wih[i] = (_Float16)Wi[i];
    for (int i = tid; i < 128 * HID;    i += 64) Whh[i] = (_Float16)Wh[i];
    for (int i = tid; i < 128;          i += 64) bsum[i] = bi[i] + bh[i];
    for (int i = tid; i < 16 * HID;     i += 64) ((_Float16*)htmp)[i] = (_Float16)0.0f;
    __syncthreads();

    const size_t b0  = (size_t)blockIdx.x * 16;
    const int m      = lane & 15;
    const int kh     = (lane >> 4) * 16;
    const int ncol   = lane & 15;
    const int rb     = (lane >> 4) * 8;     // C-tile row base for this lane
    v8f ct = {};                            // cell state tile (16 batch x 16 hid)

    for (int t = 0; t < TT; ++t) {
        v8f g[4];
        for (int gt = 0; gt < 4; ++gt) {
            float bb = bsum[gt * 32 + w * 16 + ncol];
            for (int r = 0; r < 8; ++r) g[gt][r] = bb;
        }
        // input projection: A = x_t [16 x in_dim], B = Wi rows (f16 in LDS)
        const _Float16* xrow = in_seq + ((b0 + m) * TT + t) * (size_t)in_dim;
        if (t + 1 < TT)
            __builtin_prefetch(in_seq + ((b0 + m) * TT + t + 1) * (size_t)in_dim, 0, 0);
        for (int k0 = 0; k0 < in_dim; k0 += 32) {
            v16h a = *(const v16h*)(xrow + k0 + kh);
            for (int gt = 0; gt < 4; ++gt) {
                v16h bf = *(const v16h*)(Wih + (size_t)(gt * 32 + w * 16 + ncol) * in_dim + k0 + kh);
                g[gt] = WMMA16(a, bf, g[gt]);
            }
        }
        // recurrent projection: A = h_{t-1} from LDS, K = 32 (single WMMA step)
        {
            v16h a = *(const v16h*)&htmp[m][kh];
            for (int gt = 0; gt < 4; ++gt) {
                v16h bf = *(const v16h*)(Whh + (size_t)(gt * 32 + w * 16 + ncol) * HID + kh);
                g[gt] = WMMA16(a, bf, g[gt]);
            }
        }
        __syncthreads();   // both waves finished reading htmp

        // gate math + state update (torch order i,f,g,o)
        for (int r = 0; r < 8; ++r) {
            float iv = sigm(g[0][r]);
            float fv = sigm(g[1][r]);
            float gv = tanhf(g[2][r]);
            float ov = sigm(g[3][r]);
            float c  = fv * ct[r] + iv * gv;
            ct[r] = c;
            float hv = ov * tanhf(c);
            int mm = rb + r;
            int hc = w * 16 + ncol;
            htmp[mm][hc] = (_Float16)hv;
            _Float16* orow = out_seq + ((b0 + mm) * TT + t) * (size_t)HID;
            orow[hc] = (_Float16)hv;
            if (hfinal != nullptr) {
                float* frow = hfinal + ((b0 + mm) * TT + t) * (size_t)HID;
                frow[hc] = hv;
            }
        }
        asm volatile("s_wait_dscnt 0" ::: "memory");
        __syncthreads();   // new h visible to both waves
    }
}

// ---------------------------------------------------------------------------
// Kernel 3: batchnorm statistics over axes (B, HID) per channel t
// ---------------------------------------------------------------------------
__global__ __launch_bounds__(256) void bn_stats_kernel(const float* __restrict__ hfinal,
                                                       float* __restrict__ stats) {
    const int t = blockIdx.x;
    const int tid = threadIdx.x;
    float s = 0.0f, s2 = 0.0f;
    const int n = BATCH * HID;
    for (int e = tid; e < n; e += 256) {
        int b = e >> 5, k = e & 31;
        float v = hfinal[((size_t)b * TT + t) * HID + k];
        s += v;
        s2 += v * v;
    }
    __shared__ float rs[256], rs2[256];
    rs[tid] = s; rs2[tid] = s2;
    __syncthreads();
    for (int st = 128; st > 0; st >>= 1) {
        if (tid < st) { rs[tid] += rs[tid + st]; rs2[tid] += rs2[tid + st]; }
        __syncthreads();
    }
    if (tid == 0) {
        float inv_n = 1.0f / (float)n;
        float mu = rs[0] * inv_n;
        float var = rs2[0] * inv_n - mu * mu;
        stats[t] = mu;
        stats[TT + t] = rsqrtf(var + 1e-5f);
    }
}

// ---------------------------------------------------------------------------
// Kernel 4: apply BN + FC + sigmoid.  One wave32 per (b,t) row.
// ---------------------------------------------------------------------------
__global__ __launch_bounds__(256) void bn_apply_kernel(const float* __restrict__ hfinal,
                                                       const float* __restrict__ stats,
                                                       const float* __restrict__ bn_w,
                                                       const float* __restrict__ bn_b,
                                                       const float* __restrict__ fc_w,
                                                       const float* __restrict__ fc_b,
                                                       float* __restrict__ feat,
                                                       float* __restrict__ outp) {
    const int tid  = threadIdx.x;
    const int lane = tid & 31;
    const int wv   = tid >> 5;
    const size_t row = (size_t)blockIdx.x * 8 + wv;    // row in [0, B*T)
    const int t = (int)(row & 31);
    float v = hfinal[row * HID + lane];
    float f = (v - stats[t]) * stats[TT + t] * bn_w[t] + bn_b[t];
    feat[row * HID + lane] = f;
    float p = f * fc_w[lane];
    for (int o = 16; o > 0; o >>= 1) p += __shfl_down(p, o, 32);
    if (lane == 0) outp[row] = sigm(p + fc_b[0]);
}

// ---------------------------------------------------------------------------
extern "C" void kernel_launch(void* const* d_in, const int* in_sizes, int n_in,
                              void* d_out, int out_size, void* d_ws, size_t ws_size,
                              hipStream_t stream) {
    (void)in_sizes; (void)n_in; (void)out_size; (void)ws_size;

    const float* x    = (const float*)d_in[0];
    const float* ls   = (const float*)d_in[1];
    const float* w1   = (const float*)d_in[2];
    const float* b1   = (const float*)d_in[3];
    const float* w3   = (const float*)d_in[4];
    const float* b3   = (const float*)d_in[5];
    const float* Wi0  = (const float*)d_in[6];
    const float* Wh0  = (const float*)d_in[7];
    const float* bi0  = (const float*)d_in[8];
    const float* bh0  = (const float*)d_in[9];
    const float* Wi   = (const float*)d_in[10];  // [4,128,32]
    const float* Wh   = (const float*)d_in[11];  // [4,128,32]
    const float* bi   = (const float*)d_in[12];  // [4,128]
    const float* bh   = (const float*)d_in[13];  // [4,128]
    const float* bn_w = (const float*)d_in[14];
    const float* bn_b = (const float*)d_in[15];
    const float* fc_w = (const float*)d_in[16];
    const float* fc_b = (const float*)d_in[17];

    float* feat = (float*)d_out;                                 // [B,T,HID]
    float* outp = (float*)d_out + (size_t)BATCH * TT * HID;      // [B,T]

    char* wsb = (char*)d_ws;
    const size_t MB = 1024 * 1024;
    float*    adjt  = (float*)(wsb);                 // 4 KB
    _Float16* w1T   = (_Float16*)(wsb + 8192);       // 64 KB
    _Float16* w3T   = (_Float16*)(wsb + 8192 + 65536);
    _Float16* hbuf  = (_Float16*)(wsb + 1 * MB);     // B*T*128 f16 = 32 MB
    _Float16* seqA  = (_Float16*)(wsb + 35 * MB);    // B*T*32 f16 = 8 MB
    _Float16* seqB  = (_Float16*)(wsb + 44 * MB);    // 8 MB
    float*    hfin  = (float*)(wsb + 53 * MB);       // B*T*32 f32 = 16 MB
    float*    stats = (float*)(wsb + 70 * MB);       // 64 floats

    adjt_kernel<<<1, 32, 0, stream>>>(adjt);
    prep_w_kernel<<<(DD * 128 + 255) / 256, 256, 0, stream>>>(w1, w3, w1T, w3T);
    gcn_kernel<<<BATCH, 256, 0, stream>>>(x, ls, b1, b3, adjt, w1T, w3T, hbuf);

    // LSTM stack: layer 0 (in=128), layers 1..4 (in=32)
    lstm_kernel<<<BATCH / 16, 64, 0, stream>>>(hbuf, seqA, Wi0, Wh0, bi0, bh0, 128, nullptr);
    _Float16* cur = seqA; _Float16* nxt = seqB;
    for (int l = 0; l < 4; ++l) {
        const float* Wil = Wi + (size_t)l * 128 * HID;
        const float* Whl = Wh + (size_t)l * 128 * HID;
        const float* bil = bi + (size_t)l * 128;
        const float* bhl = bh + (size_t)l * 128;
        float* hf = (l == 3) ? hfin : nullptr;
        lstm_kernel<<<BATCH / 16, 64, 0, stream>>>(cur, nxt, Wil, Whl, bil, bhl, HID, hf);
        _Float16* tmp = cur; cur = nxt; nxt = tmp;
    }

    bn_stats_kernel<<<TT, 256, 0, stream>>>(hfin, stats);
    bn_apply_kernel<<<(BATCH * TT) / 8, 256, 0, stream>>>(hfin, stats, bn_w, bn_b,
                                                          fc_w, fc_b, feat, outp);
}
